// MultiHeadAttention_16681652978374
// MI455X (gfx1250) — matrix operand order repaired and checked by reading, not compiled, not run on hardware
//
#include <hip/hip_runtime.h>
#include <hip/hip_bf16.h>

// ---------------------------------------------------------------------------
// MI455X (gfx1250) fused multi-head attention, bf16 WMMA + fp32 accumulate.
//
// Sizes (fixed by the reference):
//   B=2, S=2048, D=1024, H=16, DK=64.
// Outputs: out  [B,S,D]        fp32  (4,194,304 elems)
//          attn [B,H,S,S]      fp32  (134,217,728 elems)  -- dominates HBM
//
// Design: the 512MB attn write is the roofline term; everything else is kept
// in LDS (16x2048 score stripe per workgroup, fits in CDNA5's 320KB WGP LDS)
// or L2 (all K/V = 16MB << 192MB L2). attn is written exactly once with
// non-temporal stores. All matmuls use v_wmma_f32_16x16x32_bf16.
// ---------------------------------------------------------------------------

typedef __attribute__((ext_vector_type(16))) __bf16 v16bf;
typedef __attribute__((ext_vector_type(8)))  __bf16 v8bf;
typedef __attribute__((ext_vector_type(8)))  float  v8f;

union V16 { v16bf v; v8bf h[2]; };

static __device__ inline v8f vzero8() {
  v8f z = {0.f, 0.f, 0.f, 0.f, 0.f, 0.f, 0.f, 0.f};
  return z;
}

static __device__ inline v8f wmma_bf16(v16bf a, v16bf b, v8f c) {
  // D = A(16x32 bf16) x B(32x16 bf16) + C(16x16 f32)
  return __builtin_amdgcn_wmma_f32_16x16x32_bf16(
      false, a, false, b, (short)0, c, false, false);
}

#define BB      2
#define SS      2048
#define DD      1024
#define HH      16
#define DKK     64
#define MTOT    (BB * SS)          // 4096 rows of activations

// ---------------------------------------------------------------------------
// Cast x (fp32) -> bf16, elementwise.
// ---------------------------------------------------------------------------
__global__ __launch_bounds__(256) void k_cast_x(const float* __restrict__ x,
                                                __bf16* __restrict__ xb, int n) {
  int i = blockIdx.x * 256 + threadIdx.x;
  if (i < n) xb[i] = (__bf16)x[i];
}

// ---------------------------------------------------------------------------
// Cast + transpose weight: wt[k][n] = w[n][k] * scale   (1024x1024)
// so that GEMM B-operand rows (K) are contiguous along N.
// Block (32,8), 32x32 tile via LDS.
// ---------------------------------------------------------------------------
__global__ __launch_bounds__(256) void k_cast_w_t(const float* __restrict__ w,
                                                  __bf16* __restrict__ wt,
                                                  float scale) {
  __shared__ float t[32][33];
  const int n0 = blockIdx.x * 32;   // output-feature dim of W
  const int k0 = blockIdx.y * 32;   // input-feature dim of W
  const int tx = threadIdx.x, ty = threadIdx.y;
  for (int i = 0; i < 32; i += 8)
    t[ty + i][tx] = w[(size_t)(n0 + ty + i) * DD + k0 + tx];
  __syncthreads();
  for (int i = 0; i < 32; i += 8)
    wt[(size_t)(k0 + ty + i) * DD + n0 + tx] = (__bf16)(t[tx][ty + i] * scale);
}

// ---------------------------------------------------------------------------
// Transpose K per head: kt[b][h][d][s] = kp[b*S+s][h*64+d]
// so score-GEMM B rows (d) are contiguous along kv.
// grid (S/32, DK/32, B*H), block (32,8).
// ---------------------------------------------------------------------------
__global__ __launch_bounds__(256) void k_transpose_k(const __bf16* __restrict__ kp,
                                                     __bf16* __restrict__ kt) {
  __shared__ __bf16 t[32][33];
  const int bh = blockIdx.z;                 // b*16 + h
  const int b = bh >> 4, h = bh & 15;
  const int s0 = blockIdx.x * 32;
  const int d0 = blockIdx.y * 32;
  const int tx = threadIdx.x, ty = threadIdx.y;
  for (int i = 0; i < 32; i += 8)
    t[ty + i][tx] = kp[(size_t)(b * SS + s0 + ty + i) * DD + h * DKK + d0 + tx];
  __syncthreads();
  for (int i = 0; i < 32; i += 8)
    kt[(size_t)(bh * DKK + d0 + ty + i) * SS + s0 + tx] = t[tx][ty + i];
}

// ---------------------------------------------------------------------------
// bf16 GEMM: C[M,N] = A[M,K] * B[K,N], A/B bf16, C bf16 or f32.
// One wave computes a 16x64 strip (4 WMMA accumulators). 8 waves / block.
// A fragment layout (16-bit A, 16x32): lane%16 = M row, lane/16 selects which
// K-octets land in the low/high halves of the 8 VGPRs (per ISA 7.12.2).
// B fragment layout (32x16): lane%16 = N col, 16 K values per lane in the same K order as A.
// Weights (2MB bf16 each) are L2-resident, so no LDS staging is needed.
// ---------------------------------------------------------------------------
__global__ __launch_bounds__(256) void k_gemm_bf16(const __bf16* __restrict__ A,
                                                   const __bf16* __restrict__ B,
                                                   void* __restrict__ C,
                                                   int M, int N, int K, int f32out) {
  const int tid  = threadIdx.x;
  const int lane = tid & 31;
  const int wave = tid >> 5;
  const int tilesN   = N >> 6;                       // 64-wide wave tiles
  const int waveTile = blockIdx.x * 8 + wave;
  const int m0 = (waveTile / tilesN) * 16;
  const int n0 = (waveTile % tilesN) * 64;

  const int arow = m0 + (lane & 15);
  const int ksel = (lane >> 4) * 8;                  // 0 or 8

  v8f acc0 = vzero8(), acc1 = vzero8(), acc2 = vzero8(), acc3 = vzero8();

  for (int k0 = 0; k0 < K; k0 += 32) {
    V16 a;
    const __bf16* ap = A + (size_t)arow * K + k0 + ksel;
    a.h[0] = *(const v8bf*)(ap);                     // K = k0 + ksel + 0..7
    a.h[1] = *(const v8bf*)(ap + 16);                // K = k0 + 16 + ksel + 0..7

    const __bf16* bp = B + (size_t)k0 * N + n0 + (lane & 15);   // lane%16 = N col
    v16bf b0, b1, b2, b3;
    for (int s = 0; s < 16; ++s) {
      const int kk = (s < 8) ? (ksel + s) : (16 + ksel + s - 8);
      const __bf16* q = bp + (size_t)kk * N;
      b0[s] = q[0]; b1[s] = q[16]; b2[s] = q[32]; b3[s] = q[48];
    }

    acc0 = wmma_bf16(a.v, b0, acc0);
    acc1 = wmma_bf16(a.v, b1, acc1);
    acc2 = wmma_bf16(a.v, b2, acc2);
    acc3 = wmma_bf16(a.v, b3, acc3);
  }

  // C/D layout: VGPR r -> M = r + 8*(lane/16), N = lane%16
  const int crow = m0 + (lane >> 4) * 8;
  const int ccol = n0 + (lane & 15);
  if (f32out) {
    float* cp = (float*)C;
    for (int r = 0; r < 8; ++r) {
      size_t base = (size_t)(crow + r) * N + ccol;
      cp[base +  0] = acc0[r];
      cp[base + 16] = acc1[r];
      cp[base + 32] = acc2[r];
      cp[base + 48] = acc3[r];
    }
  } else {
    __bf16* cp = (__bf16*)C;
    for (int r = 0; r < 8; ++r) {
      size_t base = (size_t)(crow + r) * N + ccol;
      cp[base +  0] = (__bf16)acc0[r];
      cp[base + 16] = (__bf16)acc1[r];
      cp[base + 32] = (__bf16)acc2[r];
      cp[base + 48] = (__bf16)acc3[r];
    }
  }
}

// ---------------------------------------------------------------------------
// Fused attention: one workgroup (8 waves) per (b, h, 16-row q-tile).
//  phase 1: S = Q Ktile^T (WMMA) + mask  -> LDS stripe [16][2048] fp32
//  phase 2: row softmax (wave32 shuffle reductions), NT-store attn (512MB,
//           the roofline term -- written exactly once)
//  phase 3: ctx = P @ V (WMMA, P re-quantized bf16 from LDS), 2-way kv-split
//           partial reduce in LDS, bf16 ctx out.
// LDS: 16*2052*4 (padded pitch to spread banks) + 8*256*4 = 139,520 B < 320KB.
// ---------------------------------------------------------------------------
#define SC_LD 2052   // padded row pitch (floats); 2052 % 64 == 4 -> bank spread

__global__ __launch_bounds__(256) void k_attention(const __bf16* __restrict__ qp,
                                                   const __bf16* __restrict__ kt,
                                                   const __bf16* __restrict__ vp,
                                                   const int* __restrict__ mask,
                                                   float* __restrict__ attn_out,
                                                   __bf16* __restrict__ ctx) {
  extern __shared__ char smem_raw[];
  float* sc   = (float*)smem_raw;          // [16][SC_LD]
  float* part = sc + 16 * SC_LD;           // [8][16*16]

  const int tid = threadIdx.x, lane = tid & 31, wave = tid >> 5;
  const int blk = blockIdx.x;              // 0 .. B*H*(S/16)-1 = 4095
  const int b  = blk >> 11;
  const int h  = (blk >> 7) & 15;
  const int q0 = (blk & 127) << 4;

  const int mrow = lane & 15;
  const int ksel = (lane >> 4) * 8;

  // ---- Q fragments: 16x64 = two 16x32 A tiles (scale already folded in) ----
  const __bf16* qrow = qp + (size_t)(b * SS + q0 + mrow) * DD + h * DKK;
  V16 a0, a1;
  a0.h[0] = *(const v8bf*)(qrow +  0 + ksel);
  a0.h[1] = *(const v8bf*)(qrow + 16 + ksel);
  a1.h[0] = *(const v8bf*)(qrow + 32 + ksel);
  a1.h[1] = *(const v8bf*)(qrow + 48 + ksel);

  // ---- phase 1: scores, wave owns kv columns [wave*256, wave*256+256) ----
  const __bf16* ktb = kt + (size_t)((b * HH + h) * DKK) * SS;
  for (int t = 0; t < 16; ++t) {
    const int n0 = wave * 256 + t * 16;
    v16bf b0, b1;
    for (int s = 0; s < 16; ++s) {
      const int kk = (s < 8) ? (ksel + s) : (16 + ksel + s - 8);
      const __bf16* q = ktb + (size_t)kk * SS + n0 + (lane & 15);
      b0[s] = q[0];
      b1[s] = q[(size_t)32 * SS];
    }
    v8f acc = vzero8();
    acc = wmma_bf16(a0.v, b0, acc);
    acc = wmma_bf16(a1.v, b1, acc);
    const int col = n0 + (lane & 15);
    const int mv  = mask[b * SS + col];          // truthy -> masked (-1e9)
    const int rb  = (lane >> 4) * 8;
    for (int r = 0; r < 8; ++r)
      sc[(rb + r) * SC_LD + col] = mv ? -1e9f : acc[r];
  }
  __syncthreads();

  // ---- phase 2: softmax, wave owns rows 2*wave and 2*wave+1 ----
  for (int ri = 0; ri < 2; ++ri) {
    const int rr = wave * 2 + ri;
    float* srow = sc + rr * SC_LD;
    float mx = -3.4e38f;
    for (int c = lane; c < SS; c += 32) mx = fmaxf(mx, srow[c]);
    for (int off = 16; off; off >>= 1) mx = fmaxf(mx, __shfl_xor(mx, off, 32));
    float sum = 0.f;
    for (int c = lane; c < SS; c += 32) {
      float e = __expf(srow[c] - mx);
      srow[c] = e;
      sum += e;
    }
    for (int off = 16; off; off >>= 1) sum += __shfl_xor(sum, off, 32);
    const float inv = 1.0f / sum;
    float* arow = attn_out +
        ((size_t)(b * HH + h) * SS + (q0 + rr)) * (size_t)SS;
    for (int c = lane; c < SS; c += 32) {
      float p = srow[c] * inv;
      srow[c] = p;
      __builtin_nontemporal_store(p, arow + c);   // attn written once, NT
    }
  }
  __syncthreads();

  // ---- phase 3: ctx = P @ V; wave -> (d-chunk = wave&3, kv-half = wave>>2) ----
  const int dc = wave & 3, kh = wave >> 2;
  const int d0 = dc * 16;
  const int kvbase = kh * 1024;
  v8f acc = vzero8();
  const __bf16* vbase = vp + (size_t)(b * SS + kvbase) * DD + h * DKK + d0;
  for (int kv0 = 0; kv0 < 1024; kv0 += 32) {
    V16 a;
    const float* pr = sc + mrow * SC_LD + kvbase + kv0 + ksel;
    for (int j = 0; j < 8; ++j) {
      a.v[j]     = (__bf16)pr[j];        // K = kv0 + ksel + j
      a.v[8 + j] = (__bf16)pr[16 + j];   // K = kv0 + 16 + ksel + j
    }
    v16bf bf;
    for (int s = 0; s < 16; ++s) {
      const int kk = (s < 8) ? (ksel + s) : (16 + ksel + s - 8);
      bf[s] = vbase[(size_t)(kv0 + kk) * DD + (lane & 15)];
    }
    acc = wmma_bf16(a.v, bf, acc);
  }
  {
    const int rb = (lane >> 4) * 8, cn = lane & 15;
    float* pw = part + wave * 256;
    for (int r = 0; r < 8; ++r) pw[(rb + r) * 16 + cn] = acc[r];
  }
  __syncthreads();
  if (wave < 4) {
    const int rb = (lane >> 4) * 8, cn = lane & 15;
    const float* p0 = part + wave * 256;
    const float* p1 = part + (wave + 4) * 256;
    for (int r = 0; r < 8; ++r) {
      const int idx = (rb + r) * 16 + cn;
      const float v = p0[idx] + p1[idx];
      ctx[(size_t)(b * SS + q0 + rb + r) * DD + h * DKK + wave * 16 + cn] =
          (__bf16)v;
    }
  }
}

// ---------------------------------------------------------------------------
// Host-side orchestration. Workspace layout (bytes, all regions 2MB-aligned):
//   xb   8MB | wtq 2MB | wtk 2MB | wtv 2MB | wto 2MB |
//   qp   8MB | kp  8MB | vp  8MB | kt  8MB | ctx 8MB   => 56MB total
// ---------------------------------------------------------------------------
extern "C" void kernel_launch(void* const* d_in, const int* in_sizes, int n_in,
                              void* d_out, int out_size, void* d_ws, size_t ws_size,
                              hipStream_t stream) {
  const float* x    = (const float*)d_in[0];
  const int*   mask = (const int*)d_in[1];
  const float* wq   = (const float*)d_in[2];
  const float* wk   = (const float*)d_in[3];
  const float* wv   = (const float*)d_in[4];
  const float* wo   = (const float*)d_in[5];
  float* out = (float*)d_out;

  char* ws = (char*)d_ws;
  const size_t MB = 1024 * 1024;
  __bf16* xb  = (__bf16*)(ws + 0);
  __bf16* wtq = (__bf16*)(ws +  8 * MB);
  __bf16* wtk = (__bf16*)(ws + 10 * MB);
  __bf16* wtv = (__bf16*)(ws + 12 * MB);
  __bf16* wto = (__bf16*)(ws + 14 * MB);
  __bf16* qp  = (__bf16*)(ws + 16 * MB);
  __bf16* kp  = (__bf16*)(ws + 24 * MB);
  __bf16* vp  = (__bf16*)(ws + 32 * MB);
  __bf16* kt  = (__bf16*)(ws + 40 * MB);
  __bf16* ctx = (__bf16*)(ws + 48 * MB);

  const int nAct = MTOT * DD;                       // 4,194,304

  // 1) casts (fold softmax scale 1/sqrt(64)=0.125 into Wq)
  k_cast_x<<<(nAct + 255) / 256, 256, 0, stream>>>(x, xb, nAct);
  dim3 tb(32, 8);
  k_cast_w_t<<<dim3(32, 32), tb, 0, stream>>>(wq, wtq, 0.125f);
  k_cast_w_t<<<dim3(32, 32), tb, 0, stream>>>(wk, wtk, 1.0f);
  k_cast_w_t<<<dim3(32, 32), tb, 0, stream>>>(wv, wtv, 1.0f);
  k_cast_w_t<<<dim3(32, 32), tb, 0, stream>>>(wo, wto, 1.0f);

  // 2) QKV projections: (4096x1024) x (1024x1024); 4096 wave-tiles / 8 = 512 WGs
  k_gemm_bf16<<<512, 256, 0, stream>>>(xb, wtq, (void*)qp, MTOT, DD, DD, 0);
  k_gemm_bf16<<<512, 256, 0, stream>>>(xb, wtk, (void*)kp, MTOT, DD, DD, 0);
  k_gemm_bf16<<<512, 256, 0, stream>>>(xb, wtv, (void*)vp, MTOT, DD, DD, 0);

  // 3) K -> [b,h,d,s]
  k_transpose_k<<<dim3(SS / 32, DKK / 32, BB * HH), tb, 0, stream>>>(kp, kt);

  // 4) fused attention; attn region starts after `out` (4,194,304 floats)
  const size_t smem = (16 * SC_LD + 8 * 256) * sizeof(float);   // 139,520 B
  k_attention<<<BB * HH * (SS / 16), 256, smem, stream>>>(
      qp, kt, vp, mask, out + (size_t)nAct, ctx);

  // 5) output projection, fp32 result straight into d_out
  k_gemm_bf16<<<512, 256, 0, stream>>>(ctx, wto, (void*)out, MTOT, DD, DD, 1);
}
